// Decoder_22917945491720
// MI455X (gfx1250) — compile-verified, matrix operand order checked by reading, not run on hardware
//
#include <hip/hip_runtime.h>

// MI455X / gfx1250 implementation of the attention-decoder forward pass.
// Memory/latency-bound workload: elementwise/conv kernels are straightforward
// VALU; the attention GEMMs (q@k^T and att@v, per (b,head)) use
// v_wmma_f32_16x16x32_f16 with f16 operands staged through LDS and f32 accum.

typedef __attribute__((ext_vector_type(16))) _Float16 v16h;
typedef __attribute__((ext_vector_type(8)))  float    v8f;

#define TPB 256

static inline int cdiv(int a, int b) { return (a + b - 1) / b; }

// ---------------------------------------------------------------------------
// generic conv kernels (grouped, optional bias, optional fused PReLU)
// ---------------------------------------------------------------------------
__global__ __launch_bounds__(TPB)
void k_conv1x1(const float* __restrict__ in, const float* __restrict__ w,
               const float* __restrict__ bias, const float* __restrict__ slope,
               float* __restrict__ out, int B, int Cin, int Cout, int HW,
               int groups, int accumulate)
{
  int idx = blockIdx.x * blockDim.x + threadIdx.x;
  int total = B * Cout * HW;
  if (idx >= total) return;
  int p  = idx % HW;
  int co = (idx / HW) % Cout;
  int b  = idx / (HW * Cout);
  int cog = Cout / groups, cig = Cin / groups;
  int g = co / cog;
  const float* ip = in + ((size_t)b * Cin + (size_t)g * cig) * HW + p;
  const float* wp = w + (size_t)co * cig;
  float s = bias ? bias[co] : 0.f;
  for (int ci = 0; ci < cig; ++ci) s += wp[ci] * ip[(size_t)ci * HW];
  if (slope) s = (s > 0.f) ? s : slope[co] * s;
  if (accumulate) out[idx] += s; else out[idx] = s;
}

__global__ __launch_bounds__(TPB)
void k_conv3x3(const float* __restrict__ in, const float* __restrict__ w,
               const float* __restrict__ bias, const float* __restrict__ slope,
               float* __restrict__ out, int B, int Cin, int Cout, int H, int W,
               int groups)
{
  int idx = blockIdx.x * blockDim.x + threadIdx.x;
  int total = B * Cout * H * W;
  if (idx >= total) return;
  int x  = idx % W;
  int y  = (idx / W) % H;
  int co = (idx / (W * H)) % Cout;
  int b  = idx / (W * H * Cout);
  int cog = Cout / groups, cig = Cin / groups;
  int g = co / cog;
  const float* ib = in + ((size_t)b * Cin + (size_t)g * cig) * H * W;
  const float* wp = w + (size_t)co * cig * 9;
  float s = bias ? bias[co] : 0.f;
  for (int ci = 0; ci < cig; ++ci) {
    const float* ic = ib + (size_t)ci * H * W;
    const float* wc = wp + ci * 9;
    for (int ky = 0; ky < 3; ++ky) {
      int iy = y + ky - 1;
      if (iy < 0 || iy >= H) continue;
      for (int kx = 0; kx < 3; ++kx) {
        int ix = x + kx - 1;
        if (ix < 0 || ix >= W) continue;
        s += wc[ky * 3 + kx] * ic[iy * W + ix];
      }
    }
  }
  if (slope) s = (s > 0.f) ? s : slope[co] * s;
  out[idx] = s;
}

// ---------------------------------------------------------------------------
// group-norm (in-place), optional fused preceding ReLU
// one block per (b, group), two passes
// ---------------------------------------------------------------------------
__global__ __launch_bounds__(1024)
void k_gnorm(float* __restrict__ x, const float* __restrict__ gamma,
             const float* __restrict__ beta, int B, int C, int HW, int G,
             int preRelu)
{
  const int bg = blockIdx.x;
  const int g = bg % G, b = bg / G;
  const int cpg = C / G;
  const int N = cpg * HW;
  float* base = x + ((size_t)b * C + (size_t)g * cpg) * HW;
  __shared__ float s1[1024], s2[1024];
  float a = 0.f, a2 = 0.f;
  for (int i = threadIdx.x; i < N; i += blockDim.x) {
    float v = base[i];
    if (preRelu) v = fmaxf(v, 0.f);
    a += v; a2 += v * v;
  }
  s1[threadIdx.x] = a; s2[threadIdx.x] = a2;
  __syncthreads();
  for (int st = blockDim.x >> 1; st > 0; st >>= 1) {
    if ((int)threadIdx.x < st) {
      s1[threadIdx.x] += s1[threadIdx.x + st];
      s2[threadIdx.x] += s2[threadIdx.x + st];
    }
    __syncthreads();
  }
  const float mu  = s1[0] / (float)N;
  const float var = s2[0] / (float)N - mu * mu;
  const float rs  = rsqrtf(var + 1e-5f);
  for (int i = threadIdx.x; i < N; i += blockDim.x) {
    float v = base[i];
    if (preRelu) v = fmaxf(v, 0.f);
    int ch = g * cpg + i / HW;
    base[i] = (v - mu) * rs * gamma[ch] + beta[ch];
  }
}

// ---------------------------------------------------------------------------
// 3x3 stride-1 pad-1 pooling (max, or avg with count_include_pad -> /9)
// ---------------------------------------------------------------------------
__global__ __launch_bounds__(TPB)
void k_pool3(const float* __restrict__ in, float* __restrict__ out,
             int B, int C, int H, int W, int isMax)
{
  int idx = blockIdx.x * blockDim.x + threadIdx.x;
  int total = B * C * H * W;
  if (idx >= total) return;
  int x = idx % W, y = (idx / W) % H;
  const float* p = in + (size_t)(idx / (W * H)) * H * W;
  if (isMax) {
    float m = -3.4e38f;
    for (int ky = -1; ky <= 1; ++ky) {
      int iy = y + ky; if (iy < 0 || iy >= H) continue;
      for (int kx = -1; kx <= 1; ++kx) {
        int ix = x + kx; if (ix < 0 || ix >= W) continue;
        m = fmaxf(m, p[iy * W + ix]);
      }
    }
    out[idx] = m;
  } else {
    float s = 0.f;
    for (int ky = -1; ky <= 1; ++ky) {
      int iy = y + ky; if (iy < 0 || iy >= H) continue;
      for (int kx = -1; kx <= 1; ++kx) {
        int ix = x + kx; if (ix < 0 || ix >= W) continue;
        s += p[iy * W + ix];
      }
    }
    out[idx] = s * (1.f / 9.f);
  }
}

__global__ __launch_bounds__(TPB)
void k_add(float* __restrict__ a, const float* __restrict__ b, int n)
{
  int idx = blockIdx.x * blockDim.x + threadIdx.x;
  if (idx < n) a[idx] += b[idx];
}

// ---------------------------------------------------------------------------
// bilinear upsample (half-pixel centers) -> write into concat buffer slice
// ---------------------------------------------------------------------------
__global__ __launch_bounds__(TPB)
void k_upsample(const float* __restrict__ in, float* __restrict__ out,
                int B, int H, int W, int S, int Ctot, int cOff)
{
  int idx = blockIdx.x * blockDim.x + threadIdx.x;
  int total = B * 16 * S * S;
  if (idx >= total) return;
  int ox = idx % S, oy = (idx / S) % S;
  int c = (idx / (S * S)) % 16, b = idx / (S * S * 16);
  float sy = (oy + 0.5f) * (float)H / (float)S - 0.5f;
  float sx = (ox + 0.5f) * (float)W / (float)S - 0.5f;
  sy = fminf(fmaxf(sy, 0.f), (float)H - 1.f);
  sx = fminf(fmaxf(sx, 0.f), (float)W - 1.f);
  int y0 = (int)sy, x0 = (int)sx;
  int y1 = min(y0 + 1, H - 1), x1 = min(x0 + 1, W - 1);
  float fy = sy - (float)y0, fx = sx - (float)x0;
  const float* p = in + ((size_t)b * 16 + c) * H * W;
  float v = p[y0 * W + x0] * (1.f - fy) * (1.f - fx)
          + p[y0 * W + x1] * (1.f - fy) * fx
          + p[y1 * W + x0] * fy * (1.f - fx)
          + p[y1 * W + x1] * fy * fx;
  out[(((size_t)b * Ctot + cOff + c) * S + oy) * S + ox] = v;
}

// ---------------------------------------------------------------------------
// SE block helpers
// ---------------------------------------------------------------------------
__global__ __launch_bounds__(TPB)
void k_mean(const float* __restrict__ xc, float* __restrict__ mean, int SS)
{
  const float* p = xc + (size_t)blockIdx.x * SS;
  __shared__ float s[TPB];
  float a = 0.f;
  for (int i = threadIdx.x; i < SS; i += TPB) a += p[i];
  s[threadIdx.x] = a; __syncthreads();
  for (int st = TPB / 2; st > 0; st >>= 1) {
    if ((int)threadIdx.x < st) s[threadIdx.x] += s[threadIdx.x + st];
    __syncthreads();
  }
  if (threadIdx.x == 0) mean[blockIdx.x] = s[0] / (float)SS;
}

__global__ __launch_bounds__(128)
void k_se(const float* __restrict__ mean, const float* __restrict__ w1,
          const float* __restrict__ w2, float* __restrict__ yse)
{
  int b = blockIdx.x;
  __shared__ float s[128];
  float a = 0.f;
  if (threadIdx.x < 80) a = w1[threadIdx.x] * mean[b * 80 + threadIdx.x];
  s[threadIdx.x] = a; __syncthreads();
  for (int st = 64; st > 0; st >>= 1) {
    if ((int)threadIdx.x < st) s[threadIdx.x] += s[threadIdx.x + st];
    __syncthreads();
  }
  float t = fmaxf(s[0], 0.f);
  if (threadIdx.x < 80)
    yse[b * 80 + threadIdx.x] = 1.f / (1.f + __expf(-w2[threadIdx.x] * t));
}

__global__ __launch_bounds__(TPB)
void k_scale(float* __restrict__ xc, const float* __restrict__ yse,
             int B, int Ctot, int SS)
{
  int idx = blockIdx.x * blockDim.x + threadIdx.x;
  int total = B * Ctot * SS;
  if (idx >= total) return;
  int c = (idx / SS) % Ctot;
  int b = idx / (SS * Ctot);
  xc[idx] *= (1.f + yse[b * Ctot + c]);
}

// ---------------------------------------------------------------------------
// WMMA patch attention: one workgroup (8 wave32) per (batch, head).
//   S = q @ k^T    (pn x pn, K = d streamed in 32-chunks, f16 ops, f32 accum)
//   att = softmax(S * pn^-0.5)
//   O = att @ v    (v = (q+k)/2, d streamed in 128-col slices, 16 cols per wave)
// Fragment packing per CDNA5 ISA 7.12.2 (16-bit A 16x32 layout; C/D 16x16).
// ---------------------------------------------------------------------------
__device__ __forceinline__ int qk_off(int p, int dd, int a, int ps, int hh,
                                      int ww, int hw, int heads, int hd,
                                      int H, int W)
{
  int pi = p / ps, pj = p % ps;
  int cc = dd / hw, rem = dd % hw;
  int ih = rem / ww, iw = rem % ww;
  int ch = (cc * 2 + a) * heads + hd;
  return (ch * H + pi * hh + ih) * W + pj * ww + iw;
}

__device__ __forceinline__ v8f wmma_qk(_Float16 (*qs)[32], _Float16 (*ks)[32],
                                       int tr, int tc, int lane, v8f acc)
{
  const int m = lane & 15, hf = lane >> 4;
  v16h a, b;
#pragma unroll
  for (int j = 0; j < 8; ++j) {
    const int k0 = ((j & 4) ? 16 : 0) + hf * 8 + (j & 3) * 2;
    a[2 * j]     = qs[tr * 16 + m][k0];
    a[2 * j + 1] = qs[tr * 16 + m][k0 + 1];
    b[2 * j]     = ks[tc * 16 + m][k0];
    b[2 * j + 1] = ks[tc * 16 + m][k0 + 1];
  }
  return __builtin_amdgcn_wmma_f32_16x16x32_f16(false, a, false, b, (short)0,
                                                acc, false, false);
}

__global__ __launch_bounds__(TPB)
void k_attention(const float* __restrict__ qkv, float* __restrict__ out,
                 int B, int heads, int H, int W, int ps)
{
  const int c = 16;
  const int blk = blockIdx.x;
  const int b = blk / heads, hd = blk % heads;
  const int hh = H / ps, ww = W / ps;
  const int hw = hh * ww;
  const int d = c * hw;
  const int pn = ps * ps;
  const int pnPad = (pn < 16) ? 16 : pn;
  const int pt = pnPad >> 4;     // 1 or 4 row/col tiles
  const int nT = pt * pt;        // 1 or 16 tiles of S
  const int CO = 2 * c * heads;
  const int tid = threadIdx.x, lane = tid & 31, wv = tid >> 5;
  const float* base = qkv + (size_t)b * CO * H * W;

  __shared__ float    S[64][65];
  __shared__ _Float16 qs[64][32];
  __shared__ _Float16 ks[64][32];
  __shared__ _Float16 attH[64][64];
  __shared__ _Float16 vC[64][130];

  // ---- Phase 1: S = q @ k^T, K streamed in 32-chunks ----
  v8f acc0, acc1;
#pragma unroll
  for (int i = 0; i < 8; ++i) { acc0[i] = 0.f; acc1[i] = 0.f; }
  const int t0 = wv, t1 = wv + 8;
  for (int dk = 0; dk < d; dk += 32) {
    for (int e = tid; e < pnPad * 32; e += TPB) {
      int p = e >> 5, dd = e & 31;
      float qv = 0.f, kv = 0.f;
      if (p < pn) {
        qv = base[qk_off(p, dk + dd, 0, ps, hh, ww, hw, heads, hd, H, W)];
        kv = base[qk_off(p, dk + dd, 1, ps, hh, ww, hw, heads, hd, H, W)];
      }
      qs[p][dd] = (_Float16)qv;
      ks[p][dd] = (_Float16)kv;
    }
    __syncthreads();
    if (t0 < nT) acc0 = wmma_qk(qs, ks, t0 / pt, t0 % pt, lane, acc0);
    if (t1 < nT) acc1 = wmma_qk(qs, ks, t1 / pt, t1 % pt, lane, acc1);
    __syncthreads();
  }
  {
    const int m = lane & 15, hf = lane >> 4;
    if (t0 < nT) {
      int tr = t0 / pt, tc = t0 % pt;
#pragma unroll
      for (int r = 0; r < 8; ++r) S[tr * 16 + r + 8 * hf][tc * 16 + m] = acc0[r];
    }
    if (t1 < nT) {
      int tr = t1 / pt, tc = t1 % pt;
#pragma unroll
      for (int r = 0; r < 8; ++r) S[tr * 16 + r + 8 * hf][tc * 16 + m] = acc1[r];
    }
  }
  __syncthreads();

  // ---- Phase 2: row softmax -> attH (f16, zero-padded to 64x64) ----
  for (int e = tid; e < 64 * 64; e += TPB) attH[e >> 6][e & 63] = (_Float16)0.f;
  __syncthreads();
  if (tid < pn) {
    const float scale = rsqrtf((float)pn);
    float mx = -3.4e38f;
    for (int j = 0; j < pn; ++j) {
      float v = S[tid][j] * scale; S[tid][j] = v; mx = fmaxf(mx, v);
    }
    float sum = 0.f;
    for (int j = 0; j < pn; ++j) {
      float v = __expf(S[tid][j] - mx); S[tid][j] = v; sum += v;
    }
    float inv = 1.f / sum;
    for (int j = 0; j < pn; ++j) attH[tid][j] = (_Float16)(S[tid][j] * inv);
  }
  __syncthreads();

  // ---- Phase 3: O = att @ v, 128 d-columns per iteration (16/wave) ----
  float* outBase = out + (size_t)b * (c * heads) * H * W;
  const int kLim = (pn == 64) ? 64 : 32;
  const int vRows = kLim;                  // rows of v actually consumed
  for (int dB = 0; dB < d; dB += 128) {
    for (int e = tid; e < vRows * 128; e += TPB) {
      int p = e >> 7, j = e & 127;
      float v = 0.f;
      if (p < pn) {
        int dd = dB + j;
        v = 0.5f * (base[qk_off(p, dd, 0, ps, hh, ww, hw, heads, hd, H, W)] +
                    base[qk_off(p, dd, 1, ps, hh, ww, hw, heads, hd, H, W)]);
      }
      vC[p][j] = (_Float16)v;
    }
    __syncthreads();
    const int sliceLocal = wv * 16;
    const int m = lane & 15, hf = lane >> 4;
    for (int rt = 0; rt < pt; ++rt) {
      v8f acc;
#pragma unroll
      for (int i = 0; i < 8; ++i) acc[i] = 0.f;
      for (int kc = 0; kc < kLim; kc += 32) {
        v16h a, bb;
#pragma unroll
        for (int j = 0; j < 8; ++j) {
          const int k0 = ((j & 4) ? 16 : 0) + hf * 8 + (j & 3) * 2;
          a[2 * j]      = attH[rt * 16 + m][kc + k0];
          a[2 * j + 1]  = attH[rt * 16 + m][kc + k0 + 1];
          bb[2 * j]     = vC[kc + k0][sliceLocal + m];
          bb[2 * j + 1] = vC[kc + k0 + 1][sliceLocal + m];
        }
        acc = __builtin_amdgcn_wmma_f32_16x16x32_f16(false, a, false, bb,
                                                     (short)0, acc, false, false);
      }
#pragma unroll
      for (int r = 0; r < 8; ++r) {
        int p = rt * 16 + r + 8 * hf;
        if (p < pn) {
          int dd = dB + sliceLocal + m;
          int pi = p / ps, pj = p % ps;
          int cc = dd / hw, rem = dd % hw;
          int ih = rem / ww, iw = rem % ww;
          int ch = cc * heads + hd;
          outBase[(ch * H + pi * hh + ih) * W + pj * ww + iw] = acc[r];
        }
      }
    }
    __syncthreads();
  }
}

// ---------------------------------------------------------------------------
// host orchestration
// ---------------------------------------------------------------------------
static void run_attn(const float* xic, void* const* d_in, int baseIdx,
                     int heads, int ps, int og, int isMax, int accumulate,
                     int H, float* qkvBuf, float* attnO, float* t1, float* t2,
                     float* yacc, hipStream_t stream)
{
  // attn param block, sorted-key order:
  // +0 o1_a +1 o1_beta +2 o1_g +3 o1_w +4 o2_a +5 o2_w
  // +6 qkv_b +7 qkv_beta +8 qkv_g +9 qkv_w
  auto F = [&](int off) { return (const float*)d_in[baseIdx + off]; };
  const int B = 4, HW = H * H;
  const int CO = 32 * heads;

  int total = B * CO * HW;
  k_conv3x3<<<cdiv(total, TPB), TPB, 0, stream>>>(
      xic, F(9), F(6), nullptr, qkvBuf, B, 16, CO, H, H, 4);
  k_gnorm<<<B * 2, 1024, 0, stream>>>(qkvBuf, F(8), F(7), B, CO, HW, 2, 1);

  k_attention<<<B * heads, TPB, 0, stream>>>(qkvBuf, attnO, B, heads, H, H, ps);

  total = B * 16 * HW;
  k_conv1x1<<<cdiv(total, TPB), TPB, 0, stream>>>(
      attnO, F(3), nullptr, F(0), t1, B, 16 * heads, 16, HW, og, 0);
  k_gnorm<<<B, 1024, 0, stream>>>(t1, F(2), F(1), B, 16, HW, 1, 0);
  k_pool3<<<cdiv(total, TPB), TPB, 0, stream>>>(t1, t2, B, 16, H, H, isMax);
  k_conv1x1<<<cdiv(total, TPB), TPB, 0, stream>>>(
      t2, F(5), nullptr, F(4), yacc, B, 16, 16, HW, og, accumulate);
}

extern "C" void kernel_launch(void* const* d_in, const int* in_sizes, int n_in,
                              void* d_out, int out_size, void* d_ws, size_t ws_size,
                              hipStream_t stream)
{
  (void)in_sizes; (void)n_in; (void)out_size; (void)ws_size;
  const int B = 4, S = 256, SS = S * S;
  auto P = [&](int i) { return (const float*)d_in[i]; };

  // workspace layout (floats)
  float* ws  = (float*)d_ws;
  float* xc  = ws;                         // 4*80*256*256 = 20,971,520
  float* r1  = ws + 20971520;              // 16,777,216
  float* r2  = ws + 37748736;              // 16,777,216
  // stage scratch carved from r1/r2
  float* qkvB  = r1;                       // <= 8,388,608
  float* attnO = r1 + 8388608;             // <= 4,194,304
  float* xic   = r1 + 12582912;            // <= 1,048,576
  float* yacc  = r1 + 13631488;            // <= 1,048,576
  float* t1    = r2;                       // <= 1,048,576
  float* t2    = r2 + 1048576;             // <= 1,048,576
  float* meanB = r2 + 2097152;             // 320
  float* yse   = meanB + 512;              // 320

  // flattened param indices (jax pytree: dict keys sorted)
  struct StageCfg { int cin, H, heads, pl, pg, b_ic, b_lsa, b_gsa; };
  const StageCfg st[5] = {
      { 64, 128, 4, 8,  4,  25,  29,  15},
      { 64,  64, 4, 8,  4,  49,  53,  39},
      {128,  32, 4, 4,  2,  73,  77,  63},
      {256,  16, 3, 2, -1,  87,  91,  -1},
      {512,   8, 2, 2, -1, 101, 105,  -1}};

  for (int s = 0; s < 5; ++s) {
    const int H = st[s].H, HW = H * H;
    const int bi = st[s].b_ic;   // +0 a, +1 b, +2 g, +3 w
    int total = B * 16 * HW;
    k_conv1x1<<<cdiv(total, TPB), TPB, 0, stream>>>(
        P(s), P(bi + 3), nullptr, P(bi + 0), xic, B, st[s].cin, 16, HW, 2, 0);
    k_gnorm<<<B, 1024, 0, stream>>>(xic, P(bi + 2), P(bi + 1), B, 16, HW, 1, 0);

    run_attn(xic, d_in, st[s].b_lsa, st[s].heads, st[s].pl, 2, /*max*/1,
             /*acc*/0, H, qkvB, attnO, t1, t2, yacc, stream);
    if (st[s].pg > 0)
      run_attn(xic, d_in, st[s].b_gsa, st[s].heads, st[s].pg, 4, /*avg*/0,
               /*acc*/1, H, qkvB, attnO, t1, t2, yacc, stream);

    k_add<<<cdiv(total, TPB), TPB, 0, stream>>>(yacc, xic, total);
    k_upsample<<<cdiv(B * 16 * SS, TPB), TPB, 0, stream>>>(
        yacc, xc, B, H, H, S, 80, s * 16);
  }

  // SE block: yse = sigmoid(w2 * relu(w1 . mean(xc)));  xc *= (1 + yse)
  k_mean<<<B * 80, TPB, 0, stream>>>(xc, meanB, SS);
  k_se<<<B, 128, 0, stream>>>(meanB, P(115), P(116), yse);
  k_scale<<<cdiv(B * 80 * SS, TPB), TPB, 0, stream>>>(xc, yse, B, 80, SS);

  // output head:  out params: 5 a1, 6 a2, 7 a3, 8 b1, 9 b2, 10 g1, 11 g2,
  //               12 w1(64,20,3,3), 13 w2(64,64), 14 w3(1,64)
  float* z1 = r1;   // 4*64*256*256 = 16,777,216 floats
  float* z2 = r2;
  int tot = B * 64 * SS;
  k_conv3x3<<<cdiv(tot, TPB), TPB, 0, stream>>>(
      xc, P(12), nullptr, P(5), z1, B, 80, 64, S, S, 4);
  k_gnorm<<<B, 1024, 0, stream>>>(z1, P(10), P(8), B, 64, SS, 1, 0);
  k_conv1x1<<<cdiv(tot, TPB), TPB, 0, stream>>>(
      z1, P(13), nullptr, P(6), z2, B, 64, 64, SS, 1, 0);
  k_gnorm<<<B, 1024, 0, stream>>>(z2, P(11), P(9), B, 64, SS, 1, 0);
  int toto = B * 1 * SS;
  k_conv1x1<<<cdiv(toto, TPB), TPB, 0, stream>>>(
      z2, P(14), nullptr, P(7), (float*)d_out, B, 64, 1, SS, 1, 0);
}
